// TransformerEncoderLayer_62929860821055
// MI455X (gfx1250) — compile-verified, matrix operand order checked by reading, not run on hardware
//
#include <hip/hip_runtime.h>
#include <stddef.h>

// ---------------------------------------------------------------------------
// Types for CDNA5 WMMA (wave32): v_wmma_f32_16x16x32_bf16
// ---------------------------------------------------------------------------
typedef __attribute__((ext_vector_type(16))) __bf16        v16bf;
typedef __attribute__((ext_vector_type(8)))  float         v8f;
typedef __attribute__((ext_vector_type(8)))  unsigned int  v8u;
typedef unsigned short u16b;

#if defined(__has_builtin)
#  if __has_builtin(__builtin_amdgcn_tensor_load_to_lds)
#    define USE_TDM 1
#  endif
#endif
#ifndef USE_TDM
#  define USE_TDM 0
#endif

static __device__ inline u16b f2bf(float f) {
    unsigned int u = __builtin_bit_cast(unsigned int, f);
    u += 0x7FFFu + ((u >> 16) & 1u);       // round-to-nearest-even truncation
    return (u16b)(u >> 16);
}

static __device__ inline v8f wmma_bf16(v16bf a, v16bf b, v8f c) {
    // (neg_a, A, neg_b, B, c_mod, C, reuse_a, reuse_b)
    return __builtin_amdgcn_wmma_f32_16x16x32_bf16(false, a, false, b, (short)0, c,
                                                   false, false);
}

// Load one 16x32 bf16 A/B fragment from LDS laid out [row][k] with row stride
// `ldr` (u16 elements).  Per ISA 7.12.2 (16-bit A 16x32): lanes 0-15 hold row
// M=lane, K-octet {0..7,16..23}; lanes 16-31 hold row M=lane-16, K-octet
// {8..15,24..31}.  B uses the identical pattern with "row" = N, so both A and
// B tiles are staged [seq-or-N][K] and read with this one helper.
static __device__ inline v16bf ldsFrag(const u16b* base, int row, int ldr, int ks) {
    int lane = threadIdx.x & 31;
    int r    = row + (lane & 15);
    int hk   = ks + ((lane >> 4) << 3);          // +0 or +8 within the K-step
    const u16b* p = base + r * ldr + hk;
    uint4 lo = *reinterpret_cast<const uint4*>(p);        // K = hk .. hk+7
    uint4 hi = *reinterpret_cast<const uint4*>(p + 16);   // K = hk+16 .. hk+23
    v8u u;
    u[0] = lo.x; u[1] = lo.y; u[2] = lo.z; u[3] = lo.w;
    u[4] = hi.x; u[5] = hi.y; u[6] = hi.z; u[7] = hi.w;
    return __builtin_bit_cast(v16bf, u);
}

#if USE_TDM
// ---------------------------------------------------------------------------
// Tensor Data Mover: one wave-level DMA stages a whole 2D bf16 tile into LDS,
// with hardware padding landing rows on the bank-conflict-free stride.
// Descriptor layout per CDNA5 ISA ch.8 (group0 128b, group1 256b).
// ---------------------------------------------------------------------------
typedef __attribute__((ext_vector_type(4))) unsigned int u32x4;
typedef __attribute__((ext_vector_type(8))) int          i32x8;
typedef __attribute__((ext_vector_type(4))) int          i32x4;

static __device__ inline void tdm_load_tile_2d(const u16b* gptr,
                                               unsigned int ldsAddr,
                                               unsigned int tileW,   // elems (dim0)
                                               unsigned int tileH,   // rows  (dim1)
                                               unsigned int strideElems,
                                               unsigned int padInterval, // enc
                                               unsigned int padAmount) { // enc
    unsigned long long ga = (unsigned long long)(size_t)gptr;
    u32x4 g0;
    g0[0] = 1u;                                           // count=1, user mode
    g0[1] = ldsAddr;                                      // lds_addr [63:32]
    g0[2] = (unsigned int)(ga & 0xFFFFFFFFu);             // global_addr lo
    g0[3] = (unsigned int)((ga >> 32) & 0x1FFFFFFu)       // global_addr[56:32]
          | (2u << 30);                                   // type = 2 (image)
    const unsigned int td0 = 0x7FFFFFFFu;                 // huge: no OOB clip
    const unsigned int td1 = 0x7FFFFFFFu;
    i32x8 g1;
    g1[0] = (int)((1u << 16)                              // data_size = 2B
                | (1u << 20)                              // pad_enable
                | (padInterval << 22)
                | (padAmount << 25));
    g1[1] = (int)((td0 & 0xFFFFu) << 16);                 // tensor_dim0 lo16
    g1[2] = (int)(((td0 >> 16) & 0xFFFFu) | ((td1 & 0xFFFFu) << 16));
    g1[3] = (int)(((td1 >> 16) & 0xFFFFu) | ((tileW & 0xFFFFu) << 16));
    g1[4] = (int)(tileH & 0xFFFFu);                       // tile_dim1, dim2=0
    g1[5] = (int)strideElems;                             // dim0_stride lo32
    g1[6] = 0;                                            // hi16 + dim1_stride
    g1[7] = 0;
    i32x4 z4 = {0, 0, 0, 0};
#if __clang_major__ >= 23
    i32x8 z8 = {0, 0, 0, 0, 0, 0, 0, 0};
    __builtin_amdgcn_tensor_load_to_lds(g0, g1, z4, z4, z8, 0);
#else
    __builtin_amdgcn_tensor_load_to_lds(g0, g1, z4, z4, 0);
#endif
}
#endif // USE_TDM

// ---------------------------------------------------------------------------
// fp32 -> bf16 conversion (grid-stride)
// ---------------------------------------------------------------------------
__global__ __launch_bounds__(256) void k_f32_to_bf16(const float* __restrict__ in,
                                                     u16b* __restrict__ out,
                                                     size_t n) {
    size_t i      = (size_t)blockIdx.x * 256 + threadIdx.x;
    size_t stride = (size_t)gridDim.x * 256;
    for (; i < n; i += stride) out[i] = f2bf(in[i]);
}

// ---------------------------------------------------------------------------
// bf16 WMMA GEMM:  C[M,N] = A[M,K] * B[K,N]  (+bias, +relu, out f32 or bf16)
// Block tile 128x128, K-stage 64 (two WMMA K-steps per barrier), 8 waves as
// 4(M)x2(N); each wave owns 32x64 = 2x4 accumulators -> 16 WMMA per stage.
// A tile staged by the Tensor Data Mover (padded rows), B tile transposed
// [n][k] by the threads so one fragment helper serves both operands.
// ---------------------------------------------------------------------------
template <bool BIAS, bool RELU, bool OUTBF16>
__global__ __launch_bounds__(256) void k_gemm_bf16(const u16b* __restrict__ A,
                                                   const u16b* __restrict__ B,
                                                   const float* __restrict__ bias,
                                                   void* __restrict__ Cout,
                                                   int M, int N, int K) {
    constexpr int LDT = 72;                 // 64 + 8 pad (u16 elems, 144 B row)
    __shared__ u16b sA[128 * LDT];
    __shared__ u16b sB[128 * LDT];

    const int tid  = threadIdx.x;
    const int wid  = tid >> 5;
    const int lane = tid & 31;
    const int wm   = wid & 3;               // 4 waves over M
    const int wn   = wid >> 2;              // 2 waves over N
    const int m0   = blockIdx.y * 128;
    const int n0   = blockIdx.x * 128;

    v8f acc[2][4] = {};

    for (int k0 = 0; k0 < K; k0 += 64) {
        __syncthreads();
#if USE_TDM
        if (wid == 0) {
            // 128 rows x 64 bf16 from A(m0, k0); pad_interval=4 -> pad after
            // every 32 DW (one 128B row); pad_amount=3 -> 4 DW = 8 elems,
            // so rows land on the LDT=72 stride.
            tdm_load_tile_2d(A + (size_t)m0 * K + k0,
                             (unsigned int)(size_t)sA,
                             64u, 128u, (unsigned int)K, 4u, 3u);
        }
#else
#pragma unroll
        for (int it = 0; it < 4; ++it) {               // 1024 uint4, coalesced
            int idx = it * 256 + tid;
            int r   = idx >> 3;
            int c8  = (idx & 7) << 3;
            uint4 d = *reinterpret_cast<const uint4*>(
                A + (size_t)(m0 + r) * K + k0 + c8);
            *reinterpret_cast<uint4*>(sA + r * LDT + c8) = d;
        }
#endif
        // B tile 64x128 transposed to [n][k]; reads coalesce along N.
        __builtin_prefetch(B + (size_t)(k0 + 64) * N + n0, 0, 1);
#pragma unroll
        for (int it = 0; it < 32; ++it) {
            int e  = it * 256 + tid;
            int kk = e >> 7;
            int nn = e & 127;
            sB[nn * LDT + kk] = B[(size_t)(k0 + kk) * N + n0 + nn];
        }
#if USE_TDM
        if (wid == 0) __builtin_amdgcn_s_wait_tensorcnt(0);
#endif
        __syncthreads();

#pragma unroll
        for (int ks = 0; ks < 64; ks += 32) {
            v16bf af0 = ldsFrag(sA, wm * 32 + 0,  LDT, ks);
            v16bf af1 = ldsFrag(sA, wm * 32 + 16, LDT, ks);
#pragma unroll
            for (int nt = 0; nt < 4; ++nt) {
                v16bf bfr = ldsFrag(sB, wn * 64 + nt * 16, LDT, ks);
                acc[0][nt] = wmma_bf16(af0, bfr, acc[0][nt]);
                acc[1][nt] = wmma_bf16(af1, bfr, acc[1][nt]);
            }
        }
    }

    // epilogue: C layout = lane half selects M+8, lane&15 selects N
    const int hf  = lane >> 4;
    const int nlo = lane & 15;
#pragma unroll
    for (int mt = 0; mt < 2; ++mt)
#pragma unroll
        for (int nt = 0; nt < 4; ++nt)
#pragma unroll
            for (int g = 0; g < 8; ++g) {
                int row = m0 + wm * 32 + mt * 16 + g + 8 * hf;
                int col = n0 + wn * 64 + nt * 16 + nlo;
                float v = acc[mt][nt][g];
                if (BIAS) v += bias[col];
                if (RELU) v = v > 0.0f ? v : 0.0f;
                if (OUTBF16)
                    ((u16b*)Cout)[(size_t)row * N + col] = f2bf(v);
                else
                    ((float*)Cout)[(size_t)row * N + col] = v;
            }
}

// ---------------------------------------------------------------------------
// Flash attention (bf16 WMMA, online softmax).  Q/K/V stored (b, t, h*64) as
// bf16.  One block = one (b,h) x 64-query tile; 4 waves, each wave owns 16
// queries.  KV streamed in 64-chunks; dk = 64 -> two K-steps per GEMM.
// ---------------------------------------------------------------------------
__global__ __launch_bounds__(128) void k_attention(const u16b* __restrict__ Q,
                                                   const u16b* __restrict__ K,
                                                   const u16b* __restrict__ V,
                                                   u16b* __restrict__ Ctx) {
    constexpr int T = 2048, E = 1024, D = 64, LD = 72;  // 72 = 64 + 8 pad
    __shared__ u16b sQ[64 * LD];
    __shared__ u16b sK[64 * LD];
    __shared__ u16b sVt[64 * LD];   // V transposed: [d][kv]
    __shared__ u16b sP[64 * LD];    // softmaxed probs for C->A relayout

    const int bh   = blockIdx.x;            // 64 = 4 batches * 16 heads
    const int b    = bh >> 4;
    const int h    = bh & 15;
    const int q0   = blockIdx.y * 64;
    const int tid  = threadIdx.x;
    const int wid  = tid >> 5;
    const int lane = tid & 31;
    const size_t base = (size_t)b * T * E + (size_t)h * D;

#pragma unroll
    for (int it = 0; it < 4; ++it) {        // stage Q 64x64, coalesced uint4
        int idx = it * 128 + tid;
        int r   = idx >> 3;
        int c8  = (idx & 7) << 3;
        uint4 d = *reinterpret_cast<const uint4*>(Q + base + (size_t)(q0 + r) * E + c8);
        *reinterpret_cast<uint4*>(sQ + r * LD + c8) = d;
    }

    v8f   o[4] = {};
    float mrow[8], lrow[8];
#pragma unroll
    for (int g = 0; g < 8; ++g) { mrow[g] = -1e30f; lrow[g] = 0.0f; }

    for (int kv0 = 0; kv0 < T; kv0 += 64) {
        __syncthreads();
#pragma unroll
        for (int it = 0; it < 4; ++it) {                  // K chunk [kv][d]
            int idx = it * 128 + tid;
            int r   = idx >> 3;
            int c8  = (idx & 7) << 3;
            uint4 d = *reinterpret_cast<const uint4*>(
                K + base + (size_t)(kv0 + r) * E + c8);
            *reinterpret_cast<uint4*>(sK + r * LD + c8) = d;
        }
#pragma unroll
        for (int it = 0; it < 32; ++it) {                 // V chunk -> [d][kv]
            int e = it * 128 + tid;
            int r = e >> 6;                               // kv
            int c = e & 63;                               // d (coalesced)
            sVt[c * LD + r] = V[base + (size_t)(kv0 + r) * E + c];
        }
        __syncthreads();

        // S = Q * K^T  (A rows = queries, B rows = kv positions)
        v8f s[4] = {};
#pragma unroll
        for (int ks = 0; ks < D; ks += 32) {
            v16bf aq = ldsFrag(sQ, wid * 16, LD, ks);
#pragma unroll
            for (int nt = 0; nt < 4; ++nt)
                s[nt] = wmma_bf16(aq, ldsFrag(sK, nt * 16, LD, ks), s[nt]);
        }
#pragma unroll
        for (int nt = 0; nt < 4; ++nt)
#pragma unroll
            for (int g = 0; g < 8; ++g) s[nt][g] *= 0.125f;   // 1/sqrt(64)

        // online softmax: rows live in 16-lane halves; xor-reduce within half
#pragma unroll
        for (int g = 0; g < 8; ++g) {
            float mx = fmaxf(fmaxf(s[0][g], s[1][g]), fmaxf(s[2][g], s[3][g]));
#pragma unroll
            for (int off = 8; off; off >>= 1)
                mx = fmaxf(mx, __shfl_xor(mx, off, 32));
            float mnew  = fmaxf(mrow[g], mx);
            float alpha = __expf(mrow[g] - mnew);
            float rs = 0.0f;
#pragma unroll
            for (int nt = 0; nt < 4; ++nt) {
                float p = __expf(s[nt][g] - mnew);
                s[nt][g] = p;
                rs += p;
            }
#pragma unroll
            for (int off = 8; off; off >>= 1)
                rs += __shfl_xor(rs, off, 32);
            lrow[g] = lrow[g] * alpha + rs;
            mrow[g] = mnew;
#pragma unroll
            for (int nt = 0; nt < 4; ++nt) o[nt][g] *= alpha;
        }

        // relayout P (C-layout f32) -> LDS -> A-layout bf16 fragments
#pragma unroll
        for (int nt = 0; nt < 4; ++nt)
#pragma unroll
            for (int g = 0; g < 8; ++g) {
                int r = wid * 16 + g + 8 * (lane >> 4);
                int c = nt * 16 + (lane & 15);
                sP[r * LD + c] = f2bf(s[nt][g]);
            }
        __syncthreads();

        // O += P * V   (B rows = d, cols = kv, from sVt)
#pragma unroll
        for (int ks = 0; ks < 64; ks += 32) {
            v16bf ap = ldsFrag(sP, wid * 16, LD, ks);
#pragma unroll
            for (int nt = 0; nt < 4; ++nt)
                o[nt] = wmma_bf16(ap, ldsFrag(sVt, nt * 16, LD, ks), o[nt]);
        }
    }

    // normalize and store merged ctx (b, t, h*64+d) as bf16
#pragma unroll
    for (int nt = 0; nt < 4; ++nt)
#pragma unroll
        for (int g = 0; g < 8; ++g) {
            int r = q0 + wid * 16 + g + 8 * (lane >> 4);
            int c = h * D + nt * 16 + (lane & 15);
            Ctx[((size_t)b * T + r) * E + c] = f2bf(o[nt][g] / lrow[g]);
        }
}

// ---------------------------------------------------------------------------
// residual + LayerNorm:  y = gain * ((a+res - mean)/(std + eps)) + bias
// one block per row of 1024; optional bf16 mirror for the next GEMM input.
// ---------------------------------------------------------------------------
__global__ __launch_bounds__(256) void k_residual_ln(const float* __restrict__ A,
                                                     const float* __restrict__ Res,
                                                     const float* __restrict__ gain,
                                                     const float* __restrict__ bias,
                                                     float* __restrict__ outF,
                                                     u16b* __restrict__ outB) {
    constexpr int N = 1024;
    __shared__ float red[2][8];
    const int row = blockIdx.x;
    const int tid = threadIdx.x;
    const float* a = A   + (size_t)row * N;
    const float* r = Res + (size_t)row * N;

    float vals[4], sum = 0.0f, sq = 0.0f;
#pragma unroll
    for (int i = 0; i < 4; ++i) {
        float v = a[tid + i * 256] + r[tid + i * 256];
        vals[i] = v; sum += v; sq += v * v;
    }
#pragma unroll
    for (int off = 16; off; off >>= 1) {
        sum += __shfl_xor(sum, off, 32);
        sq  += __shfl_xor(sq,  off, 32);
    }
    int wid = tid >> 5, lane = tid & 31;
    if (lane == 0) { red[0][wid] = sum; red[1][wid] = sq; }
    __syncthreads();
    if (wid == 0) {
        float s = (lane < 8) ? red[0][lane] : 0.0f;
        float q = (lane < 8) ? red[1][lane] : 0.0f;
#pragma unroll
        for (int off = 4; off; off >>= 1) {
            s += __shfl_xor(s, off, 32);
            q += __shfl_xor(q, off, 32);
        }
        if (lane == 0) { red[0][0] = s; red[1][0] = q; }
    }
    __syncthreads();
    float mean = red[0][0] / N;
    float var  = red[1][0] / N - mean * mean;
    float inv  = 1.0f / (sqrtf(fmaxf(var, 0.0f)) + 1e-5f);
#pragma unroll
    for (int i = 0; i < 4; ++i) {
        int c = tid + i * 256;
        float y = gain[c] * ((vals[i] - mean) * inv) + bias[c];
        outF[(size_t)row * N + c] = y;
        if (outB) outB[(size_t)row * N + c] = f2bf(y);
    }
}

// ---------------------------------------------------------------------------
// driver
// ---------------------------------------------------------------------------
extern "C" void kernel_launch(void* const* d_in, const int* in_sizes, int n_in,
                              void* d_out, int out_size, void* d_ws, size_t ws_size,
                              hipStream_t stream) {
    (void)in_sizes; (void)n_in; (void)out_size; (void)ws_size;
    constexpr int B = 4, T = 2048, E = 1024, FF = 4096;
    constexpr int M = B * T;                         // 8192 rows
    const size_t EE = (size_t)E * E, EF = (size_t)E * FF;
    const size_t ME = (size_t)M * E, MF = (size_t)M * FF;

    const float* x     = (const float*)d_in[0];
    const float* Wq    = (const float*)d_in[1];
    const float* Wk    = (const float*)d_in[2];
    const float* Wv    = (const float*)d_in[3];
    const float* Wo    = (const float*)d_in[4];
    const float* W1    = (const float*)d_in[5];
    const float* b1    = (const float*)d_in[6];
    const float* W2    = (const float*)d_in[7];
    const float* b2    = (const float*)d_in[8];
    const float* gain1 = (const float*)d_in[9];
    const float* bias1 = (const float*)d_in[10];
    const float* gain2 = (const float*)d_in[11];
    const float* bias2 = (const float*)d_in[12];
    float* out = (float*)d_out;

    char* cur = (char*)d_ws;
    auto carve = [&cur](size_t bytes) -> void* {
        void* p = cur;
        cur += (bytes + 255) & ~(size_t)255;
        return p;
    };
    u16b* wqb  = (u16b*)carve(EE * 2);
    u16b* wkb  = (u16b*)carve(EE * 2);
    u16b* wvb  = (u16b*)carve(EE * 2);
    u16b* wob  = (u16b*)carve(EE * 2);
    u16b* w1b  = (u16b*)carve(EF * 2);
    u16b* w2b  = (u16b*)carve(EF * 2);
    u16b* xb   = (u16b*)carve(ME * 2);
    u16b* qb   = (u16b*)carve(ME * 2);
    u16b* kb   = (u16b*)carve(ME * 2);
    u16b* vb   = (u16b*)carve(ME * 2);
    u16b* ctxb = (u16b*)carve(ME * 2);
    u16b* x1b  = (u16b*)carve(ME * 2);
    u16b* hb   = (u16b*)carve(MF * 2);
    float* attf = (float*)carve(ME * 4);   // attended; reused for FF2 output
    float* x1f  = (float*)carve(ME * 4);

    // 1) fp32 -> bf16 staging of weights and x
    k_f32_to_bf16<<<2048, 256, 0, stream>>>(Wq, wqb, EE);
    k_f32_to_bf16<<<2048, 256, 0, stream>>>(Wk, wkb, EE);
    k_f32_to_bf16<<<2048, 256, 0, stream>>>(Wv, wvb, EE);
    k_f32_to_bf16<<<2048, 256, 0, stream>>>(Wo, wob, EE);
    k_f32_to_bf16<<<4096, 256, 0, stream>>>(W1, w1b, EF);
    k_f32_to_bf16<<<4096, 256, 0, stream>>>(W2, w2b, EF);
    k_f32_to_bf16<<<4096, 256, 0, stream>>>(x,  xb,  ME);

    // 2) Q/K/V projections (bf16 WMMA, bf16 outputs for attention staging)
    dim3 gP(E / 128, M / 128);
    k_gemm_bf16<false, false, true><<<gP, 256, 0, stream>>>(xb, wqb, nullptr, qb, M, E, E);
    k_gemm_bf16<false, false, true><<<gP, 256, 0, stream>>>(xb, wkb, nullptr, kb, M, E, E);
    k_gemm_bf16<false, false, true><<<gP, 256, 0, stream>>>(xb, wvb, nullptr, vb, M, E, E);

    // 3) flash attention -> merged context (bf16)
    k_attention<<<dim3(B * 16, T / 64), 128, 0, stream>>>(qb, kb, vb, ctxb);

    // 4) output projection (f32 out for residual)
    k_gemm_bf16<false, false, false><<<gP, 256, 0, stream>>>(ctxb, wob, nullptr, attf, M, E, E);

    // 5) residual + LN1 -> x1 (f32 + bf16 mirror)
    k_residual_ln<<<M, 256, 0, stream>>>(attf, x, gain1, bias1, x1f, x1b);

    // 6) FFN: relu(x1 @ W1 + b1) -> bf16, then @ W2 + b2 -> f32
    k_gemm_bf16<true, true, true><<<dim3(FF / 128, M / 128), 256, 0, stream>>>(
        x1b, w1b, b1, hb, M, FF, E);
    k_gemm_bf16<true, false, false><<<gP, 256, 0, stream>>>(hb, w2b, b2, attf, M, E, FF);

    // 7) residual + LN2 -> output (f32)
    k_residual_ln<<<M, 256, 0, stream>>>(attf, x1f, gain2, bias2, out, nullptr);
}
